// DynamicLoRAAttention_66073776881804
// MI455X (gfx1250) — compile-verified, matrix operand order checked by reading, not run on hardware
//
#include <hip/hip_runtime.h>
#include <stdint.h>

// ---------------- problem constants ----------------
#define BATCH   2
#define TSEQ    2048
#define DIMW    1024           // DIM == INNER == HEADS*DH
#define MTOK    (BATCH*TSEQ)   // 4096
#define HEADS   16
#define DH      64
#define RLORA   8
#define LSCALE  0.25f
#define NPATCH  256

typedef _Float16 v16h  __attribute__((ext_vector_type(16)));
typedef _Float16 h8    __attribute__((ext_vector_type(8)));
typedef float    v8f   __attribute__((ext_vector_type(8)));
typedef unsigned int  u32x4 __attribute__((ext_vector_type(4)));
typedef int           i32x4 __attribute__((ext_vector_type(4)));
typedef int           i32x8 __attribute__((ext_vector_type(8)));

// ---------------- fragment load (row-major [row][k], two b128 per lane) ---
__device__ __forceinline__ v16h load_frag_rm(const _Float16* base, int ld, int lane) {
  int r  = lane & 15;
  int kb = (lane >> 4) << 3;          // 0 or 8
  const _Float16* p = base + (size_t)r * ld + kb;
  h8 lo = *(const h8*)(p);
  h8 hi = *(const h8*)(p + 16);
  return __builtin_shufflevector(lo, hi, 0,1,2,3,4,5,6,7,8,9,10,11,12,13,14,15);
}

// ---------------- TDM 2D tile load: global f16 tile -> LDS ----------------
// 6-arg builtin form: (u32x4 g0, i32x8 g1, i32x4 g2, i32x4 g3, i32x8 g4, i32 cpol)
__device__ __forceinline__ void tdm_load_2d(unsigned lds_off, const void* gptr,
                                            unsigned tdim0, unsigned tdim1,
                                            unsigned tile0, unsigned tile1,
                                            unsigned stride0 /*elements*/) {
  unsigned long long ga = (unsigned long long)(uintptr_t)gptr;
  u32x4 g0;
  g0[0] = 1u;                                        // count=1, user descriptor
  g0[1] = lds_off;                                   // LDS byte address
  g0[2] = (unsigned)(ga & 0xFFFFFFFFu);              // global_addr[31:0]
  g0[3] = (unsigned)((ga >> 32) & 0x01FFFFFFu) | (2u << 30);  // addr[56:32] | type=2
  i32x8 g1;
  g1[0] = (int)(1u << 16);                           // workgroup_mask=0, data_size=1 (2B)
  g1[1] = (int)((tdim0 & 0xFFFFu) << 16);            // tensor_dim0[15:0] @ bits 63:48
  g1[2] = (int)((tdim0 >> 16) | ((tdim1 & 0xFFFFu) << 16));
  g1[3] = (int)((tdim1 >> 16) | (tile0 << 16));      // tile_dim0
  g1[4] = (int)(tile1 & 0xFFFFu);                    // tile_dim1 (tile_dim2=0)
  g1[5] = (int)stride0;                              // tensor_dim0_stride[31:0]
  g1[6] = 0;                                         // stride0 hi16 | stride1 lo16
  g1[7] = 0;
  i32x4 z4 = {0, 0, 0, 0};
  i32x8 z8 = {0, 0, 0, 0, 0, 0, 0, 0};
  __builtin_amdgcn_tensor_load_to_lds(g0, g1, z4, z4, z8, 0);
}

// ---------------- LayerNorm -> f16 ----------------
__global__ void ln_f16_kernel(const float* __restrict__ x, const float* __restrict__ g,
                              const float* __restrict__ b, _Float16* __restrict__ out) {
  __shared__ float red[256];
  int row = blockIdx.x, t = threadIdx.x;
  const float* xr = x + (size_t)row * DIMW;
  float s = 0.f;
  for (int i = t; i < DIMW; i += 256) s += xr[i];
  red[t] = s; __syncthreads();
  for (int k = 128; k > 0; k >>= 1) { if (t < k) red[t] += red[t + k]; __syncthreads(); }
  float mu = red[0] * (1.0f / DIMW);
  __syncthreads();
  float v = 0.f;
  for (int i = t; i < DIMW; i += 256) { float d = xr[i] - mu; v += d * d; }
  red[t] = v; __syncthreads();
  for (int k = 128; k > 0; k >>= 1) { if (t < k) red[t] += red[t + k]; __syncthreads(); }
  float rstd = rsqrtf(red[0] * (1.0f / DIMW) + 1e-5f);
  for (int i = t; i < DIMW; i += 256)
    out[(size_t)row * DIMW + i] = (_Float16)((xr[i] - mu) * rstd * g[i] + b[i]);
}

// ---------------- f32 -> f16 cast ----------------
__global__ void cast_f16_kernel(const float* __restrict__ a, _Float16* __restrict__ o, int n) {
  int i = blockIdx.x * blockDim.x + threadIdx.x;
  if (i < n) o[i] = (_Float16)a[i];
}

// ---------------- 8-wide low-rank dot: out[m][r] = X[m,:]·A[r,:] ----------
__global__ void dot8_kernel(const _Float16* __restrict__ X, const float* __restrict__ A,
                            float* __restrict__ out) {
  __shared__ float red[256];
  int row = blockIdx.x, t = threadIdx.x;
  const _Float16* xr = X + (size_t)row * DIMW;
  float xv[4];
#pragma unroll
  for (int c = 0; c < 4; ++c) xv[c] = (float)xr[t + 256 * c];
  for (int r = 0; r < RLORA; ++r) {
    const float* Ar = A + (size_t)r * DIMW;
    float s = 0.f;
#pragma unroll
    for (int c = 0; c < 4; ++c) s += xv[c] * Ar[t + 256 * c];
    red[t] = s; __syncthreads();
    for (int k = 128; k > 0; k >>= 1) { if (t < k) red[t] += red[t + k]; __syncthreads(); }
    if (t == 0) out[(size_t)row * RLORA + r] = red[0];
    __syncthreads();
  }
}

__global__ void combine_kernel(const float* __restrict__ a, const float* __restrict__ g,
                               float* __restrict__ u, int n, float s) {
  int i = blockIdx.x * blockDim.x + threadIdx.x;
  if (i < n) u[i] = s * a[i] * g[i];
}

// ---------------- dynamic-LoRA GEMM: C = X·Wt + u·Bm^T ---------------------
// X [M][K] f16, W [N][K] f16 (row-major == direct B-frags), u [M][8], Bm [N][8]
// Block tile 128(M) x 64(N); 4 waves x 32 rows; TDM double-buffered W tiles.
__global__ void gemm_dynlora_kernel(const _Float16* __restrict__ X,
                                    const _Float16* __restrict__ W,
                                    const float* __restrict__ U,
                                    const float* __restrict__ Bm,
                                    _Float16* __restrict__ out16,
                                    float* __restrict__ out32,
                                    int M, int N, int K, float outScale) {
  __shared__ __align__(16) _Float16 Wtile[2][64 * 32];
  int tid = threadIdx.x, lane = tid & 31, wave = tid >> 5;
  int mBase = blockIdx.y * 128 + wave * 32;
  int nBase = blockIdx.x * 64;

  v8f acc[2][4];
#pragma unroll
  for (int mi = 0; mi < 2; ++mi)
#pragma unroll
    for (int nt = 0; nt < 4; ++nt)
#pragma unroll
      for (int i = 0; i < 8; ++i) acc[mi][nt][i] = 0.f;

  unsigned lds0 = (unsigned)(uintptr_t)(&Wtile[0][0]);
  unsigned lds1 = (unsigned)(uintptr_t)(&Wtile[1][0]);
  const _Float16* wBase = W + (size_t)nBase * K;

  int nsteps = K / 32;
  // prologue: stage tile 0
  if (wave == 0)
    tdm_load_2d(lds0, wBase, (unsigned)K, (unsigned)(N - nBase), 32u, 64u, (unsigned)K);

  for (int it = 0; it < nsteps; ++it) {
    int cur = it & 1;
    if (wave == 0) {
      if (it + 1 < nsteps) {
        // overlap: launch next tile's DMA, then wait only for the current one
        // (TENSORcnt is in-order per wave: <=1 outstanding => tile `it` done)
        tdm_load_2d(cur ? lds0 : lds1, wBase + (size_t)(it + 1) * 32,
                    (unsigned)K, (unsigned)(N - nBase), 32u, 64u, (unsigned)K);
        __builtin_amdgcn_s_wait_tensorcnt(1);
      } else {
        __builtin_amdgcn_s_wait_tensorcnt(0);
      }
    }
    __syncthreads();   // Wtile[cur] valid for all waves

    const _Float16* wc = &Wtile[cur][0];
    int kt = it * 32;
    v16h a0 = load_frag_rm(X + (size_t)mBase * K + kt, K, lane);
    v16h a1 = load_frag_rm(X + (size_t)(mBase + 16) * K + kt, K, lane);
#pragma unroll
    for (int nt = 0; nt < 4; ++nt) {
      v16h b = load_frag_rm(wc + nt * 16 * 32, 32, lane);
      acc[0][nt] = __builtin_amdgcn_wmma_f32_16x16x32_f16(false, a0, false, b,
                                                          (short)0, acc[0][nt], false, false);
      acc[1][nt] = __builtin_amdgcn_wmma_f32_16x16x32_f16(false, a1, false, b,
                                                          (short)0, acc[1][nt], false, false);
    }
    __syncthreads();   // all reads of Wtile[cur] done before it is re-targeted
  }

  // epilogue: rank-8 LoRA update + scale + store
  int nLo = lane & 15;
  int mOff = (lane >> 4) << 3;
  float br[4][8];
#pragma unroll
  for (int nt = 0; nt < 4; ++nt) {
    const float* bp = Bm + (size_t)(nBase + nt * 16 + nLo) * RLORA;
#pragma unroll
    for (int r = 0; r < 8; ++r) br[nt][r] = bp[r];
  }
#pragma unroll
  for (int mi = 0; mi < 2; ++mi) {
#pragma unroll
    for (int j = 0; j < 8; ++j) {
      int m = mBase + mi * 16 + mOff + j;
      const float* up = U + (size_t)m * RLORA;
      float ur[8];
#pragma unroll
      for (int r = 0; r < 8; ++r) ur[r] = up[r];
#pragma unroll
      for (int nt = 0; nt < 4; ++nt) {
        int n = nBase + nt * 16 + nLo;
        float v = acc[mi][nt][j];
#pragma unroll
        for (int r = 0; r < 8; ++r) v += ur[r] * br[nt][r];
        v *= outScale;
        if (out16) out16[(size_t)m * N + n] = (_Float16)v;
        if (out32) out32[(size_t)m * N + n] = v;
      }
    }
  }
}

// ---------------- flash attention, block-causal (256-token frames) --------
// Q pre-scaled by DH^-0.5. Layout: [tok][h*64+dh] f16.
__global__ void attn_kernel(const _Float16* __restrict__ Q,
                            const _Float16* __restrict__ Kx,
                            const _Float16* __restrict__ V,
                            _Float16* __restrict__ O) {
  __shared__ __align__(16) _Float16 Kt[32 * 64];   // [j_local][dh]
  __shared__ __align__(16) _Float16 Vt[64 * 32];   // [dh][j_local]
  __shared__ __align__(16) _Float16 Pl[4 * 16 * 32];
  int tid = threadIdx.x, lane = tid & 31, wave = tid >> 5;
  int iTile = blockIdx.x;   // 32 tiles of 64 rows
  int h = blockIdx.y;
  int b = blockIdx.z;
  size_t tokBase = (size_t)b * TSEQ;
  const _Float16* qb = Q + tokBase * DIMW + h * DH;
  const _Float16* kb = Kx + tokBase * DIMW + h * DH;
  const _Float16* vb = V + tokBase * DIMW + h * DH;
  int r0 = iTile * 64 + wave * 16;

  v16h aq0 = load_frag_rm(qb + (size_t)r0 * DIMW, DIMW, lane);
  v16h aq1 = load_frag_rm(qb + (size_t)r0 * DIMW + 32, DIMW, lane);

  int jmax = ((iTile * 64) / NPATCH + 1) * NPATCH;  // frame-causal prefix

  float mI[8], lI[8];
  v8f oacc[4];
#pragma unroll
  for (int v = 0; v < 8; ++v) { mI[v] = -3.0e38f; lI[v] = 0.f; }
#pragma unroll
  for (int d = 0; d < 4; ++d)
#pragma unroll
    for (int i = 0; i < 8; ++i) oacc[d][i] = 0.f;

  _Float16* Pw = &Pl[wave * 16 * 32];
  int mRow = (lane >> 4) << 3;
  int nLo = lane & 15;

  for (int j0 = 0; j0 < jmax; j0 += 32) {
    __syncthreads();
    // cooperative stage: K row-major, V transposed
    for (int c = tid; c < 256; c += 128) {
      int row = c >> 3, col = (c & 7) << 3;
      h8 kv = *(const h8*)(kb + (size_t)(j0 + row) * DIMW + col);
      *(h8*)(&Kt[row * 64 + col]) = kv;
      h8 vv = *(const h8*)(vb + (size_t)(j0 + row) * DIMW + col);
#pragma unroll
      for (int e = 0; e < 8; ++e) Vt[(col + e) * 32 + row] = vv[e];
    }
    __syncthreads();

    // S = Q Kt^T (16x32), f32 accum; Q pre-scaled
    v8f s[2];
#pragma unroll
    for (int nt = 0; nt < 2; ++nt) {
#pragma unroll
      for (int i = 0; i < 8; ++i) s[nt][i] = 0.f;
      v16h bk0 = load_frag_rm(&Kt[nt * 16 * 64], 64, lane);
      v16h bk1 = load_frag_rm(&Kt[nt * 16 * 64 + 32], 64, lane);
      s[nt] = __builtin_amdgcn_wmma_f32_16x16x32_f16(false, aq0, false, bk0, (short)0, s[nt], false, false);
      s[nt] = __builtin_amdgcn_wmma_f32_16x16x32_f16(false, aq1, false, bk1, (short)0, s[nt], false, false);
    }

    // online softmax (row = v + mRow, spread over 16-lane half)
    float mx[8];
#pragma unroll
    for (int v = 0; v < 8; ++v) mx[v] = fmaxf(s[0][v], s[1][v]);
    for (int msk = 1; msk < 16; msk <<= 1) {
#pragma unroll
      for (int v = 0; v < 8; ++v) mx[v] = fmaxf(mx[v], __shfl_xor(mx[v], msk, 32));
    }
    float rs[8];
#pragma unroll
    for (int v = 0; v < 8; ++v) {
      float nm = fmaxf(mI[v], mx[v]);
      float sc = __expf(mI[v] - nm);
      mI[v] = nm;
      lI[v] *= sc;
#pragma unroll
      for (int d = 0; d < 4; ++d) oacc[d][v] *= sc;
      float p0 = __expf(s[0][v] - nm);
      float p1 = __expf(s[1][v] - nm);
      s[0][v] = p0; s[1][v] = p1;
      rs[v] = p0 + p1;
    }
    for (int msk = 1; msk < 16; msk <<= 1) {
#pragma unroll
      for (int v = 0; v < 8; ++v) rs[v] += __shfl_xor(rs[v], msk, 32);
    }
#pragma unroll
    for (int v = 0; v < 8; ++v) lI[v] += rs[v];

    // P: C-layout -> LDS -> A-frag (wave-private, LDS ops in order per wave)
#pragma unroll
    for (int v = 0; v < 8; ++v) {
      int r = mRow + v;
      Pw[r * 32 + nLo]      = (_Float16)s[0][v];
      Pw[r * 32 + 16 + nLo] = (_Float16)s[1][v];
    }
    v16h pa = load_frag_rm(Pw, 32, lane);
#pragma unroll
    for (int dt = 0; dt < 4; ++dt) {
      v16h bv = load_frag_rm(&Vt[dt * 16 * 32], 32, lane);
      oacc[dt] = __builtin_amdgcn_wmma_f32_16x16x32_f16(false, pa, false, bv, (short)0, oacc[dt], false, false);
    }
  }

  // normalize + store f16
#pragma unroll
  for (int dt = 0; dt < 4; ++dt)
#pragma unroll
    for (int v = 0; v < 8; ++v) {
      int tok = r0 + mRow + v;
      float val = oacc[dt][v] / lI[v];
      O[(tokBase + tok) * DIMW + h * DH + dt * 16 + nLo] = (_Float16)val;
    }
}

// ---------------- host-side orchestration ----------------
extern "C" void kernel_launch(void* const* d_in, const int* in_sizes, int n_in,
                              void* d_out, int out_size, void* d_ws, size_t ws_size,
                              hipStream_t stream) {
  (void)in_sizes; (void)n_in; (void)out_size; (void)ws_size;
  const float* x    = (const float*)d_in[0];
  const float* mtok = (const float*)d_in[1];
  const float* ng = (const float*)d_in[2];
  const float* nb = (const float*)d_in[3];
  const float* mg = (const float*)d_in[4];
  const float* mb = (const float*)d_in[5];
  const float* Wq = (const float*)d_in[6];  const float* Aq = (const float*)d_in[7];
  const float* Bq = (const float*)d_in[8];  const float* Gq = (const float*)d_in[9];
  const float* Wk = (const float*)d_in[10]; const float* Ak = (const float*)d_in[11];
  const float* Bk = (const float*)d_in[12]; const float* Gk = (const float*)d_in[13];
  const float* Wv = (const float*)d_in[14]; const float* Av = (const float*)d_in[15];
  const float* Bv = (const float*)d_in[16]; const float* Gv = (const float*)d_in[17];
  const float* Wo = (const float*)d_in[18]; const float* Ao = (const float*)d_in[19];
  const float* Bo = (const float*)d_in[20]; const float* Go = (const float*)d_in[21];
  // d_in[22] = mask: derived analytically (frame-block causal), unused.

  char* ws = (char*)d_ws;
  size_t off = 0;
  auto take = [&](size_t bytes) -> char* {
    char* p = ws + off;
    off += (bytes + 255) & ~(size_t)255;
    return p;
  };
  const size_t actB = (size_t)MTOK * DIMW * sizeof(_Float16);   // 8 MB
  const size_t wB   = (size_t)DIMW * DIMW * sizeof(_Float16);   // 2 MB
  const size_t uB   = (size_t)MTOK * RLORA * sizeof(float);     // 128 KB

  _Float16* xn16 = (_Float16*)take(actB);
  _Float16* mn16 = (_Float16*)take(actB);
  _Float16* q16  = (_Float16*)take(actB);
  _Float16* k16  = (_Float16*)take(actB);
  _Float16* v16  = (_Float16*)take(actB);
  _Float16* Wq16 = (_Float16*)take(wB);
  _Float16* Wk16 = (_Float16*)take(wB);
  _Float16* Wv16 = (_Float16*)take(wB);
  _Float16* Wo16 = (_Float16*)take(wB);
  float* tA    = (float*)take(uB);
  float* tG    = (float*)take(uB);
  float* uq    = (float*)take(uB);
  float* uk    = (float*)take(uB);
  float* uv    = (float*)take(uB);
  float* uo    = (float*)take(uB);
  float* gateo = (float*)take(uB);
  _Float16* a16 = xn16;   // alias: xn16 dead after V projection, attn runs later

  // 1) LayerNorms -> f16
  ln_f16_kernel<<<MTOK, 256, 0, stream>>>(x, ng, nb, xn16);
  ln_f16_kernel<<<MTOK, 256, 0, stream>>>(mtok, mg, mb, mn16);

  // 2) weight casts
  int wn = DIMW * DIMW;
  cast_f16_kernel<<<(wn + 255) / 256, 256, 0, stream>>>(Wq, Wq16, wn);
  cast_f16_kernel<<<(wn + 255) / 256, 256, 0, stream>>>(Wk, Wk16, wn);
  cast_f16_kernel<<<(wn + 255) / 256, 256, 0, stream>>>(Wv, Wv16, wn);
  cast_f16_kernel<<<(wn + 255) / 256, 256, 0, stream>>>(Wo, Wo16, wn);

  // 3) low-rank gates u = LSCALE*(A·xn)*(G·mn)
  int un = MTOK * RLORA, ub = (un + 255) / 256;
  dot8_kernel<<<MTOK, 256, 0, stream>>>(xn16, Aq, tA);
  dot8_kernel<<<MTOK, 256, 0, stream>>>(mn16, Gq, tG);
  combine_kernel<<<ub, 256, 0, stream>>>(tA, tG, uq, un, LSCALE);
  dot8_kernel<<<MTOK, 256, 0, stream>>>(xn16, Ak, tA);
  dot8_kernel<<<MTOK, 256, 0, stream>>>(mn16, Gk, tG);
  combine_kernel<<<ub, 256, 0, stream>>>(tA, tG, uk, un, LSCALE);
  dot8_kernel<<<MTOK, 256, 0, stream>>>(xn16, Av, tA);
  dot8_kernel<<<MTOK, 256, 0, stream>>>(mn16, Gv, tG);
  combine_kernel<<<ub, 256, 0, stream>>>(tA, tG, uv, un, LSCALE);
  dot8_kernel<<<MTOK, 256, 0, stream>>>(mn16, Go, gateo);

  // 4) Q/K/V projections (WMMA + double-buffered TDM weights); Q pre-scaled
  dim3 gg(DIMW / 64, MTOK / 128);
  gemm_dynlora_kernel<<<gg, 128, 0, stream>>>(xn16, Wq16, uq, Bq, q16, nullptr,
                                              MTOK, DIMW, DIMW, 0.125f);
  gemm_dynlora_kernel<<<gg, 128, 0, stream>>>(xn16, Wk16, uk, Bk, k16, nullptr,
                                              MTOK, DIMW, DIMW, 1.0f);
  gemm_dynlora_kernel<<<gg, 128, 0, stream>>>(xn16, Wv16, uv, Bv, v16, nullptr,
                                              MTOK, DIMW, DIMW, 1.0f);

  // 5) flash attention with frame-block causal prefix
  attn_kernel<<<dim3(TSEQ / 64, HEADS, BATCH), 128, 0, stream>>>(q16, k16, v16, a16);

  // 6) output low-rank gate, then O-projection straight to f32 d_out
  dot8_kernel<<<MTOK, 256, 0, stream>>>(a16, Ao, tA);
  combine_kernel<<<ub, 256, 0, stream>>>(tA, gateo, uo, un, LSCALE);
  gemm_dynlora_kernel<<<gg, 128, 0, stream>>>(a16, Wo16, uo, Bo, nullptr, (float*)d_out,
                                              MTOK, DIMW, DIMW, 1.0f);
}
